// ExphormerLayer_7705171329699
// MI455X (gfx1250) — compile-verified, matrix operand order checked by reading, not run on hardware
//
#include <hip/hip_runtime.h>

#define NODES 50000
#define DMODEL 128
#define NE 800000
#define NEE 200000

typedef __attribute__((ext_vector_type(16))) _Float16 v16h;
typedef __attribute__((ext_vector_type(8)))  _Float16 v8h;
typedef __attribute__((ext_vector_type(8)))  float    v8f;

// ---------------------------------------------------------------------------
// Pack fp32 weight [K x Ncols] (row-major, as used in x @ W) into the WMMA
// B-operand layout for v_wmma_f32_16x16x32_f16:
//   P[((ntile*(K/32)+kc)*32 + lane)*16 + h]
// lane 0..15 -> n = ntile*16+lane, halves h: group g=h/8 -> K block
//   k = kc*32 + g*16 + (lane<16?0:8) + (h&7)
// ---------------------------------------------------------------------------
__global__ void pack_w_kernel(const float* __restrict__ W, _Float16* __restrict__ P,
                              int K, int Ncols) {
  int tid = blockIdx.x * blockDim.x + threadIdx.x;
  if (tid >= K * Ncols) return;
  int h     = tid & 15;
  int lane  = (tid >> 4) & 31;
  int kch   = K >> 5;
  int kc    = (tid >> 9) % kch;
  int ntile = (tid >> 9) / kch;
  int n     = ntile * 16 + (lane & 15);
  int koff  = (lane < 16) ? 0 : 8;
  int k     = kc * 32 + (h >> 3) * 16 + koff + (h & 7);
  P[tid] = (_Float16)W[(size_t)k * Ncols + n];
}

// ---------------------------------------------------------------------------
// LayerNorm over D=128: one wave per row, 4 elems/lane, butterfly reduce.
// Optional f32 and/or f16 outputs.
// ---------------------------------------------------------------------------
__global__ __launch_bounds__(256) void ln_kernel(
    const float* __restrict__ X, const float* __restrict__ G,
    const float* __restrict__ B, float* __restrict__ Of,
    _Float16* __restrict__ Oh, int rows) {
  int lane = threadIdx.x & 31;
  int row  = blockIdx.x * 8 + (threadIdx.x >> 5);
  if (row >= rows) return;
  const float* xr = X + (size_t)row * DMODEL;
  float v0 = xr[lane], v1 = xr[lane + 32], v2 = xr[lane + 64], v3 = xr[lane + 96];
  float s = v0 + v1 + v2 + v3;
#pragma unroll
  for (int o = 16; o > 0; o >>= 1) s += __shfl_xor(s, o, 32);
  float mu = s * (1.0f / DMODEL);
  float d0 = v0 - mu, d1 = v1 - mu, d2 = v2 - mu, d3 = v3 - mu;
  float q = d0 * d0 + d1 * d1 + d2 * d2 + d3 * d3;
#pragma unroll
  for (int o = 16; o > 0; o >>= 1) q += __shfl_xor(q, o, 32);
  float rstd = rsqrtf(q * (1.0f / DMODEL) + 1e-5f);
  float y0 = d0 * rstd * G[lane]      + B[lane];
  float y1 = d1 * rstd * G[lane + 32] + B[lane + 32];
  float y2 = d2 * rstd * G[lane + 64] + B[lane + 64];
  float y3 = d3 * rstd * G[lane + 96] + B[lane + 96];
  size_t base = (size_t)row * DMODEL;
  if (Of) {
    Of[base + lane] = y0; Of[base + lane + 32] = y1;
    Of[base + lane + 64] = y2; Of[base + lane + 96] = y3;
  }
  if (Oh) {
    Oh[base + lane] = (_Float16)y0; Oh[base + lane + 32] = (_Float16)y1;
    Oh[base + lane + 64] = (_Float16)y2; Oh[base + lane + 96] = (_Float16)y3;
  }
}

// ---------------------------------------------------------------------------
// GEMM: C[rows x Ncols] = A[rows x K](f16) @ Bp(packed f16) + bias (+resid)
// One wave -> one 16x16 tile via v_wmma_f32_16x16x32_f16. rows % 16 == 0.
// ---------------------------------------------------------------------------
__global__ __launch_bounds__(256) void gemm_wmma_kernel(
    const _Float16* __restrict__ A, const _Float16* __restrict__ Bp,
    const float* __restrict__ bias, const float* __restrict__ resid,
    float* __restrict__ Cf, _Float16* __restrict__ Ch,
    int rows, int K, int Ncols, int relu) {
  int lane  = threadIdx.x & 31;
  int wv    = threadIdx.x >> 5;
  int row0  = blockIdx.x * 16;
  int ntile = blockIdx.y * 8 + wv;
  if (ntile * 16 >= Ncols) return;
  int kchunks = K >> 5;
  int m    = lane & 15;
  int koff = (lane < 16) ? 0 : 8;
  const _Float16* arow = A + (size_t)(row0 + m) * K + koff;
  const _Float16* bptr = Bp + ((size_t)ntile * kchunks * 32 + lane) * 16;
  v8f acc = {};
  for (int kc = 0; kc < kchunks; ++kc) {
    v8h lo = *(const v8h*)(arow + kc * 32);
    v8h hi = *(const v8h*)(arow + kc * 32 + 16);
    v16h a;
#pragma unroll
    for (int i = 0; i < 8; ++i) { a[i] = lo[i]; a[8 + i] = hi[i]; }
    v16h b = *(const v16h*)(bptr + (size_t)kc * 32 * 16);
    acc = __builtin_amdgcn_wmma_f32_16x16x32_f16(false, a, false, b,
                                                 (short)0, acc, false, false);
  }
  int n = ntile * 16 + m;
  float bv = bias ? bias[n] : 0.0f;
  int rbase = row0 + ((lane < 16) ? 0 : 8);
#pragma unroll
  for (int r = 0; r < 8; ++r) {
    int row = rbase + r;
    float val = acc[r] + bv;
    if (relu) val = fmaxf(val, 0.0f);
    size_t idx = (size_t)row * Ncols + n;
    if (resid) val += resid[idx];
    if (Cf) Cf[idx] = val;
    if (Ch) Ch[idx] = (_Float16)val;
  }
}

// ---------------------------------------------------------------------------
// Attention scratch init: agg=0 over N*128; m=-inf(mapped), denom=0 over N*8.
// ---------------------------------------------------------------------------
__global__ void attn_init_kernel(unsigned* __restrict__ mU,
                                 float* __restrict__ denom,
                                 float* __restrict__ agg) {
  int tid = blockIdx.x * blockDim.x + threadIdx.x;
  if (tid < NODES * DMODEL) agg[tid] = 0.0f;
  if (tid < NODES * 8) { mU[tid] = 0x007FFFFFu; denom[tid] = 0.0f; }  // map(-inf)
}

__device__ __forceinline__ unsigned ord_map(float f) {
  unsigned u = __float_as_uint(f);
  return (u & 0x80000000u) ? ~u : (u | 0x80000000u);
}

// scores[e,h] = (q[dst,h] . k[src,h]) / 4 ; atomicMax into mapped-uint m.
__global__ void score_kernel(const float* __restrict__ q, const float* __restrict__ k,
                             const int* __restrict__ src, const int* __restrict__ dst,
                             float* __restrict__ scores, unsigned* __restrict__ mU,
                             int nE) {
  int tid = blockIdx.x * blockDim.x + threadIdx.x;
  if (tid >= nE * 8) return;
  int e = tid >> 3, h = tid & 7;
  int s = src[e], d = dst[e];
  const float* qp = q + (size_t)d * DMODEL + h * 16;
  const float* kp = k + (size_t)s * DMODEL + h * 16;
  float acc = 0.0f;
#pragma unroll
  for (int i = 0; i < 16; ++i) acc += qp[i] * kp[i];
  acc *= 0.25f;  // 1/sqrt(16)
  scores[tid] = acc;
  atomicMax(&mU[(size_t)d * 8 + h], ord_map(acc));
}

// unmap m; -inf (isolated node) / NaN -> 0
__global__ void mfin_kernel(float* __restrict__ mF, int n) {
  int tid = blockIdx.x * blockDim.x + threadIdx.x;
  if (tid >= n) return;
  unsigned u = ((unsigned*)mF)[tid];
  unsigned orig = (u & 0x80000000u) ? (u & 0x7FFFFFFFu) : ~u;
  float f = __uint_as_float(orig);
  if (!(f >= -3.0e38f && f <= 3.0e38f)) f = 0.0f;
  mF[tid] = f;
}

// p = exp(score - m[dst]); denom[dst,h] += p; agg[dst,h,:] += p * v[src,h,:]
__global__ void softmax_agg_kernel(const float* __restrict__ scores,
                                   const float* __restrict__ mF,
                                   const float* __restrict__ v,
                                   const int* __restrict__ src,
                                   const int* __restrict__ dst,
                                   float* __restrict__ denom,
                                   float* __restrict__ agg, int nE) {
  int tid = blockIdx.x * blockDim.x + threadIdx.x;
  if (tid >= nE * 8) return;
  int e = tid >> 3, h = tid & 7;
  int s = src[e], d = dst[e];
  float p = __expf(scores[tid] - mF[(size_t)d * 8 + h]);
  atomicAdd(&denom[(size_t)d * 8 + h], p);
  const float* vp = v + (size_t)s * DMODEL + h * 16;
  float* ap = agg + (size_t)d * DMODEL + h * 16;
#pragma unroll
  for (int i = 0; i < 16; ++i) atomicAdd(&ap[i], p * vp[i]);
}

// out_h = agg / max(denom, 1e-9), to f16 for the Wo GEMM
__global__ void attn_norm_kernel(const float* __restrict__ agg,
                                 const float* __restrict__ denom,
                                 _Float16* __restrict__ outH, int n) {
  int tid = blockIdx.x * blockDim.x + threadIdx.x;
  if (tid >= n) return;
  int node = tid >> 7;
  int h = (tid >> 4) & 7;
  float dn = denom[(size_t)node * 8 + h];
  outH[tid] = (_Float16)(agg[tid] / fmaxf(dn, 1e-9f));
}

// x1 = x + sigmoid(alpha)*xl + (1-sigmoid(alpha))*xe
__global__ void combine_kernel(const float* __restrict__ x, const float* __restrict__ xl,
                               const float* __restrict__ xe, const float* __restrict__ alpha,
                               float* __restrict__ out, int n) {
  int tid = blockIdx.x * blockDim.x + threadIdx.x;
  if (tid >= n) return;
  float a = 1.0f / (1.0f + __expf(-alpha[0]));
  out[tid] = x[tid] + a * xl[tid] + (1.0f - a) * xe[tid];
}

// ---------------------------------------------------------------------------
extern "C" void kernel_launch(void* const* d_in, const int* in_sizes, int n_in,
                              void* d_out, int out_size, void* d_ws, size_t ws_size,
                              hipStream_t stream) {
  const float* x    = (const float*)d_in[0];
  const int*   eidx = (const int*)d_in[1];
  const int*   xidx = (const int*)d_in[2];
  const float* Wsrc[8]; const float* bsrc[8];
  for (int i = 0; i < 8; ++i) {   // Wq_l,bq_l,Wk_l,bk_l,Wv_l,bv_l,Wo_l,bo_l,then _e
    Wsrc[i] = (const float*)d_in[3 + 2 * i];
    bsrc[i] = (const float*)d_in[4 + 2 * i];
  }
  const float* W1 = (const float*)d_in[19]; const float* b1 = (const float*)d_in[20];
  const float* W2 = (const float*)d_in[21]; const float* b2 = (const float*)d_in[22];
  const float* ln1g = (const float*)d_in[23]; const float* ln1b = (const float*)d_in[24];
  const float* ln2g = (const float*)d_in[25]; const float* ln2b = (const float*)d_in[26];
  const float* ln3g = (const float*)d_in[27]; const float* ln3b = (const float*)d_in[28];
  const float* alpha = (const float*)d_in[29];
  float* out = (float*)d_out;

  // ---- workspace bump allocator (phase-aliased, ~210 MB) ----
  char* ws = (char*)d_ws;
  size_t off = 0;
  auto alloc = [&](size_t bytes) -> void* {
    void* p = ws + off;
    off = (off + bytes + 255) & ~(size_t)255;
    return p;
  };
  _Float16* pw[8];
  for (int i = 0; i < 8; ++i)
    pw[i] = (_Float16*)alloc((size_t)DMODEL * DMODEL * sizeof(_Float16));
  _Float16* pW1 = (_Float16*)alloc((size_t)DMODEL * 512 * sizeof(_Float16));
  _Float16* pW2 = (_Float16*)alloc((size_t)512 * DMODEL * sizeof(_Float16));
  _Float16* xnh = (_Float16*)alloc((size_t)NODES * DMODEL * sizeof(_Float16));
  float* qf = (float*)alloc((size_t)NODES * DMODEL * 4);
  float* kf = (float*)alloc((size_t)NODES * DMODEL * 4);
  float* vf = (float*)alloc((size_t)NODES * DMODEL * 4);
  float* mBuf  = (float*)alloc((size_t)NODES * 8 * 4);
  float* denom = (float*)alloc((size_t)NODES * 8 * 4);
  char* regionX = (char*)alloc((size_t)NE * 8 * 4 + (size_t)NODES * DMODEL * 4);
  float* scores = (float*)regionX;                        // attention phases
  float* agg    = (float*)(regionX + (size_t)NE * 8 * 4); // attention phases
  _Float16* h1h = (_Float16*)regionX;                     // FFN phase (51.2 MB)
  _Float16* ah  = (_Float16*)alloc((size_t)NODES * DMODEL * sizeof(_Float16)); // attn_h / x2h
  float* xl = (float*)alloc((size_t)NODES * DMODEL * 4);
  float* xe = (float*)alloc((size_t)NODES * DMODEL * 4);
  float* x1 = qf;   // aliases: qkv dead after attention phases
  float* x2 = kf;
  float* x3 = vf;

  // ---- pack all weights into WMMA-B layout ----
  for (int i = 0; i < 8; ++i)
    pack_w_kernel<<<(DMODEL * DMODEL + 255) / 256, 256, 0, stream>>>(
        Wsrc[i], pw[i], DMODEL, DMODEL);
  pack_w_kernel<<<(DMODEL * 512 + 255) / 256, 256, 0, stream>>>(W1, pW1, DMODEL, 512);
  pack_w_kernel<<<(512 * DMODEL + 255) / 256, 256, 0, stream>>>(W2, pW2, 512, DMODEL);

  auto gemm = [&](const _Float16* A, const _Float16* Bp, const float* bias,
                  const float* resid, float* Cf, _Float16* Ch,
                  int K, int Ncols, int relu) {
    dim3 grid(NODES / 16, Ncols / 128);
    gemm_wmma_kernel<<<grid, 256, 0, stream>>>(A, Bp, bias, resid, Cf, Ch,
                                               NODES, K, Ncols, relu);
  };

  // ---- LN1 -> f16 activations ----
  ln_kernel<<<(NODES + 7) / 8, 256, 0, stream>>>(x, ln1g, ln1b, nullptr, xnh, NODES);

  auto attention = [&](int base, const int* ei, int nE, float* outX) {
    gemm(xnh, pw[base + 0], bsrc[base + 0], nullptr, qf, nullptr, 128, 128, 0);
    gemm(xnh, pw[base + 1], bsrc[base + 1], nullptr, kf, nullptr, 128, 128, 0);
    gemm(xnh, pw[base + 2], bsrc[base + 2], nullptr, vf, nullptr, 128, 128, 0);
    attn_init_kernel<<<(NODES * DMODEL + 255) / 256, 256, 0, stream>>>(
        (unsigned*)mBuf, denom, agg);
    score_kernel<<<(nE * 8 + 255) / 256, 256, 0, stream>>>(
        qf, kf, ei, ei + nE, scores, (unsigned*)mBuf, nE);
    mfin_kernel<<<(NODES * 8 + 255) / 256, 256, 0, stream>>>(mBuf, NODES * 8);
    softmax_agg_kernel<<<(nE * 8 + 255) / 256, 256, 0, stream>>>(
        scores, mBuf, vf, ei, ei + nE, denom, agg, nE);
    attn_norm_kernel<<<(NODES * DMODEL + 255) / 256, 256, 0, stream>>>(
        agg, denom, ah, NODES * DMODEL);
    gemm(ah, pw[base + 3], bsrc[base + 3], nullptr, outX, nullptr, 128, 128, 0);
  };

  attention(0, eidx, NE, xl);    // local attention
  attention(4, xidx, NEE, xe);   // expander attention

  // ---- gated residual + LN2 ----
  combine_kernel<<<(NODES * DMODEL + 255) / 256, 256, 0, stream>>>(
      x, xl, xe, alpha, x1, NODES * DMODEL);
  ln_kernel<<<(NODES + 7) / 8, 256, 0, stream>>>(x1, ln2g, ln2b, x2, ah, NODES);

  // ---- FFN (fused relu, fused residual) + LN3 -> output ----
  gemm(ah, pW1, b1, nullptr, nullptr, h1h, 128, 512, 1);   // relu, f16 out
  gemm(h1h, pW2, b2, x2, x3, nullptr, 512, 128, 0);        // + residual x2
  ln_kernel<<<(NODES + 7) / 8, 256, 0, stream>>>(x3, ln3g, ln3b, out, nullptr, NODES);
}
